// SigmoidAttention_3470333575670
// MI455X (gfx1250) — matrix operand order repaired and checked by reading, not compiled, not run on hardware
//
#include <hip/hip_runtime.h>
#include <hip/hip_bf16.h>

// ---------------------------------------------------------------------------
// Sigmoid attention for MI455X (gfx1250), bf16 WMMA path + TDM A-panel staging.
// B=4, S=2048, D=1024.  All GEMMs run through v_wmma_f32_16x16x32_bf16.
// A panels (16 x K) are DMA'd to LDS once per block via tensor_load_to_lds
// with TDM padding (16B per 512 elements) for bank-conflict-free ds_load_b128.
// ---------------------------------------------------------------------------

typedef __bf16 bf16_t;
typedef __attribute__((ext_vector_type(16))) __bf16 v16bf;
typedef __attribute__((ext_vector_type(8)))  float  v8f;
typedef __attribute__((ext_vector_type(4)))  unsigned int u32x4;
typedef __attribute__((ext_vector_type(8)))  int i32x8;
typedef __attribute__((ext_vector_type(4)))  int i32x4;

#define BATCH 4
#define SEQ   2048
#define DMODEL 1024
#define BS    (BATCH * SEQ)

// TDM pad: 4 dwords (8 bf16 elems) inserted every 256 dwords (512 bf16 elems).
// Padded chunk = 512 + 8 = 520 elements = 1040 bytes.
#define CHUNK_ELEMS 512
#define CHUNK_PAD_ELEMS 520

__host__ __device__ constexpr int lds_panel_bytes(int K) {
    return (16 * K / CHUNK_ELEMS) * (CHUNK_PAD_ELEMS * 2);
}

// ---------------------------------------------------------------------------
// TDM: DMA a 16-row x K-col bf16 panel (row stride K) from global into LDS.
// D# fields per CDNA5 ISA §8.3/8.4.  Issued by one wave; waits TENSORcnt.
// This toolchain exposes the 6-arg builtin:
//   (uint32x4 g0, int32x8 g1, int32x4 g2, int32x4 g3, int32x8 extra, i32 cpol)
// ---------------------------------------------------------------------------
__device__ __forceinline__ void tdm_load_panel(const bf16_t* __restrict__ gsrc,
                                               unsigned lds_addr, int K) {
    unsigned long long ga = (unsigned long long)(uintptr_t)gsrc;
    u32x4 g0;
    g0[0] = 1u;                                   // count=1, user descriptor
    g0[1] = lds_addr;                             // LDS byte address
    g0[2] = (unsigned)(ga & 0xFFFFFFFFu);         // global_addr[31:0]
    g0[3] = (unsigned)((ga >> 32) & 0x1FFFFFFu)   // global_addr[56:32]
          | (2u << 30);                           // type = 2 ("image")
    i32x8 g1;
    g1[0] = (1 << 16)                             // data_size = 1 -> 2 bytes
          | (1 << 20)                             // pad_enable
          | (7 << 22)                             // pad_interval: 256 dwords
          | (3 << 25);                            // pad_amount: 4 dwords
    g1[1] = (K & 0xFFFF) << 16;                   // tensor_dim0[15:0]
    g1[2] = ((K >> 16) & 0xFFFF) | (16 << 16);    // tensor_dim0[31:16] | tensor_dim1[15:0]
    g1[3] = (K & 0xFFFF) << 16;                   // tensor_dim1[31:16]=0 | tile_dim0
    g1[4] = 16;                                   // tile_dim1 = 16, tile_dim2 = 0
    g1[5] = K;                                    // tensor_dim0_stride[31:0]
    g1[6] = 0;                                    // stride hi | dim1_stride lo
    g1[7] = 0;
    const i32x4 z4 = {0, 0, 0, 0};                // groups 2/3 unused (2D tile)
    const i32x8 z8 = {0, 0, 0, 0, 0, 0, 0, 0};
    __builtin_amdgcn_tensor_load_to_lds(g0, g1, z4, z4, z8, 0);
}

// ---------------------------------------------------------------------------
// Fragment loaders, matching CDNA5 ISA §7.12.2 VGPR layouts (wave32).
// A (16x32 bf16, MxK): lane L<16 -> row L, K = {0..7, 16..23};
//                      lane L>=16 -> row L-16, K = {8..15, 24..31}.
//   A comes from the TDM-padded LDS panel (chunked addressing).
// B (32x16 bf16, KxN): lane L<16 -> col L, K = {0..7, 16..23};
//                      lane L>=16 -> col L-16, K = {8..15, 24..31}.
//   B-source matrices are stored [N][K] row-major -> two 16B loads per lane.
// ---------------------------------------------------------------------------
__device__ __forceinline__ v16bf load_a_frag_lds(const bf16_t* __restrict__ panel,
                                                 int k, int K) {
    const int lane = threadIdx.x & 31;
    const int m    = lane & 15;
    const int koff = (lane & 16) ? 8 : 0;
    const int idx  = m * K + k + koff;            // element index in panel
    // padded LDS address; a fragment never straddles a 512-elem chunk
    const bf16_t* p = panel + (idx >> 9) * CHUNK_PAD_ELEMS + (idx & 511);
    v16bf r;
    ((uint4*)&r)[0] = *(const uint4*)(p);         // K = koff .. koff+7
    ((uint4*)&r)[1] = *(const uint4*)(p + 16);    // K = koff+16 .. koff+23
    return r;
}

__device__ __forceinline__ v16bf load_b_frag(const bf16_t* __restrict__ Bn, int ldb) {
    const int lane = threadIdx.x & 31;
    const int n    = lane & 15;
    const int koff = (lane & 16) ? 8 : 0;
    const bf16_t* p = Bn + n * ldb + koff;
    v16bf r;
    ((uint4*)&r)[0] = *(const uint4*)(p);
    ((uint4*)&r)[1] = *(const uint4*)(p + 16);
    return r;
}

// One wave computes a 16(M) x 128(N) strip: 8 f32 accumulator tiles.
// A: padded LDS panel [16][K]; B-source: row-major [N][K] global (ld = K).
__device__ __forceinline__ void gemm_strip8(const bf16_t* __restrict__ panelA,
                                            const bf16_t* __restrict__ Bbase,
                                            int K, v8f acc[8]) {
    for (int k = 0; k < K; k += 32) {
        v16bf a = load_a_frag_lds(panelA, k, K);
#pragma unroll
        for (int t = 0; t < 8; ++t) {
            v16bf b = load_b_frag(Bbase + (t * 16) * K + k, K);
            acc[t] = __builtin_amdgcn_wmma_f32_16x16x32_bf16(
                false, a, false, b, (short)0, acc[t], false, false);
        }
    }
}

__device__ __forceinline__ void zero_acc(v8f acc[8]) {
    const v8f z = {0.f, 0.f, 0.f, 0.f, 0.f, 0.f, 0.f, 0.f};
#pragma unroll
    for (int t = 0; t < 8; ++t) acc[t] = z;
}

// Stage A panel via TDM and barrier the block.
__device__ __forceinline__ bf16_t* stage_a_panel(const bf16_t* __restrict__ Arow, int K) {
    extern __shared__ char smem_dyn[];
    bf16_t* panel = (bf16_t*)smem_dyn;
    if ((threadIdx.x >> 5) == 0) {                // wave 0 issues the DMA
        tdm_load_panel(Arow, (unsigned)(uintptr_t)panel, K);
        __builtin_amdgcn_s_wait_tensorcnt(0);     // only issuing wave has TENSORcnt
    }
    __syncthreads();                              // publish LDS to all waves
    return panel;
}

// ---------------------------------------------------------------------------
// fp32 -> bf16 elementwise convert
// ---------------------------------------------------------------------------
__global__ __launch_bounds__(256)
void k_cvt_bf16(const float* __restrict__ src, bf16_t* __restrict__ dst, int n) {
    for (int i = blockIdx.x * blockDim.x + threadIdx.x; i < n;
         i += gridDim.x * blockDim.x)
        dst[i] = (bf16_t)src[i];
}

// ---------------------------------------------------------------------------
// Y = LayerNorm(X @ W^T + bias) -> bf16, row-major [BS][D].
// One block: 16 rows x 1024 cols (8 waves x 128 cols) so the LN row
// reduction lives in one block.  C-tile layout: lanes 0-15 hold rows 0-7
// (vgpr r -> row r), lanes 16-31 hold rows 8-15.
// ---------------------------------------------------------------------------
__global__ __launch_bounds__(256)
void k_gemm_ln(const bf16_t* __restrict__ X, const bf16_t* __restrict__ W,
               const float* __restrict__ bias, bf16_t* __restrict__ Y) {
    const int K = DMODEL, N = DMODEL;
    const int wave = threadIdx.x >> 5;
    const int lane = threadIdx.x & 31;
    const int row0 = blockIdx.x * 16;
    const int n0   = wave * 128;
    const int mb   = (lane & 16) ? 8 : 0;

    bf16_t* panel = stage_a_panel(X + (size_t)row0 * K, K);

    v8f acc[8];
    zero_acc(acc);
    gemm_strip8(panel, W + (size_t)n0 * K, K, acc);

#pragma unroll
    for (int t = 0; t < 8; ++t) {
        float bn = bias[n0 + t * 16 + (lane & 15)];
#pragma unroll
        for (int r = 0; r < 8; ++r) acc[t][r] += bn;
    }

    // per-row partial sums over this wave's 128 columns
    float s1r[8], s2r[8];
#pragma unroll
    for (int r = 0; r < 8; ++r) {
        float s1 = 0.f, s2 = 0.f;
#pragma unroll
        for (int t = 0; t < 8; ++t) {
            float v = acc[t][r];
            s1 += v; s2 += v * v;
        }
        // butterfly reduce across the 16 lanes that share this row
        for (int off = 8; off >= 1; off >>= 1) {
            s1 += __shfl_xor(s1, off, 16);
            s2 += __shfl_xor(s2, off, 16);
        }
        s1r[r] = s1; s2r[r] = s2;
    }

    __shared__ float redS1[8][16];
    __shared__ float redS2[8][16];
    if ((lane & 15) == 0) {
#pragma unroll
        for (int r = 0; r < 8; ++r) {
            redS1[wave][mb + r] = s1r[r];
            redS2[wave][mb + r] = s2r[r];
        }
    }
    __syncthreads();

    float mu[8], rs[8];
#pragma unroll
    for (int r = 0; r < 8; ++r) {
        const int m = mb + r;
        float S1 = 0.f, S2 = 0.f;
#pragma unroll
        for (int w = 0; w < 8; ++w) { S1 += redS1[w][m]; S2 += redS2[w][m]; }
        float mean = S1 * (1.0f / (float)N);
        float var  = S2 * (1.0f / (float)N) - mean * mean;
        mu[r] = mean;
        rs[r] = rsqrtf(var + 1e-5f);
    }

#pragma unroll
    for (int t = 0; t < 8; ++t) {
        const int n = n0 + t * 16 + (lane & 15);
#pragma unroll
        for (int r = 0; r < 8; ++r) {
            float v = (acc[t][r] - mu[r]) * rs[r];
            Y[(size_t)(row0 + mb + r) * N + n] = (bf16_t)v;
        }
    }
}

// ---------------------------------------------------------------------------
// VT[b][e][t] = (X @ Wv^T + bias)[b*S+t][e]  (transposed store, bf16)
// ---------------------------------------------------------------------------
__global__ __launch_bounds__(256)
void k_gemm_vt(const bf16_t* __restrict__ X, const bf16_t* __restrict__ W,
               const float* __restrict__ bias, bf16_t* __restrict__ VT) {
    const int K = DMODEL;
    const int wave = threadIdx.x >> 5;
    const int lane = threadIdx.x & 31;
    const int row0 = blockIdx.x * 16;        // global row in [0, BS)
    const int b    = row0 >> 11;             // / SEQ
    const int sl   = row0 & (SEQ - 1);
    const int n0   = wave * 128;
    const int mb   = (lane & 16) ? 8 : 0;

    bf16_t* panel = stage_a_panel(X + (size_t)row0 * K, K);

    v8f acc[8];
    zero_acc(acc);
    gemm_strip8(panel, W + (size_t)n0 * K, K, acc);

#pragma unroll
    for (int t = 0; t < 8; ++t) {
        const int e = n0 + t * 16 + (lane & 15);
        float bn = bias[e];
        bf16_t* dst = VT + (size_t)b * DMODEL * SEQ + (size_t)e * SEQ + sl;
#pragma unroll
        for (int r = 0; r < 8; ++r)
            dst[mb + r] = (bf16_t)(acc[t][r] + bn);
    }
}

// ---------------------------------------------------------------------------
// P[s_glob][t] = sigmoid(qn[s]·kn[t]/sqrt(D) - ln S) -> bf16, [BS][SEQ]
// gridDim.y = 2 covers t in two 1024-wide halves.
// ---------------------------------------------------------------------------
__global__ __launch_bounds__(256)
void k_gemm_scores(const bf16_t* __restrict__ Q, const bf16_t* __restrict__ Kn,
                   bf16_t* __restrict__ P) {
    const int K = DMODEL;
    const int wave = threadIdx.x >> 5;
    const int lane = threadIdx.x & 31;
    const int row0 = blockIdx.x * 16;
    const int b    = row0 >> 11;
    const int n0   = blockIdx.y * 1024 + wave * 128;   // t offset within batch
    const int mb   = (lane & 16) ? 8 : 0;

    bf16_t* panel = stage_a_panel(Q + (size_t)row0 * K, K);

    v8f acc[8];
    zero_acc(acc);
    const bf16_t* Bbase = Kn + (size_t)b * SEQ * K + (size_t)n0 * K;
    gemm_strip8(panel, Bbase, K, acc);

    const float scale = 0.03125f;            // 1/sqrt(1024)
    const float lse   = 7.62461899f;         // ln(2048)
#pragma unroll
    for (int t = 0; t < 8; ++t) {
        const int tt = n0 + t * 16 + (lane & 15);
#pragma unroll
        for (int r = 0; r < 8; ++r) {
            float v = acc[t][r] * scale - lse;
            float p = 1.0f / (1.0f + __expf(-v));
            P[(size_t)(row0 + mb + r) * SEQ + tt] = (bf16_t)p;
        }
    }
}

// ---------------------------------------------------------------------------
// AO[s][e] = sum_t P[s][t] * V[t][e], using VT [b][e][t] as the [N][K] operand.
// ---------------------------------------------------------------------------
__global__ __launch_bounds__(256)
void k_gemm_pv(const bf16_t* __restrict__ P, const bf16_t* __restrict__ VT,
               bf16_t* __restrict__ AO) {
    const int K = SEQ;                       // contract over t
    const int wave = threadIdx.x >> 5;
    const int lane = threadIdx.x & 31;
    const int row0 = blockIdx.x * 16;
    const int b    = row0 >> 11;
    const int n0   = wave * 128;             // e offset
    const int mb   = (lane & 16) ? 8 : 0;

    bf16_t* panel = stage_a_panel(P + (size_t)row0 * K, K);

    v8f acc[8];
    zero_acc(acc);
    const bf16_t* Bbase = VT + (size_t)b * DMODEL * SEQ + (size_t)n0 * K;
    gemm_strip8(panel, Bbase, K, acc);

#pragma unroll
    for (int t = 0; t < 8; ++t) {
        const int e = n0 + t * 16 + (lane & 15);
#pragma unroll
        for (int r = 0; r < 8; ++r)
            AO[(size_t)(row0 + mb + r) * DMODEL + e] = (bf16_t)acc[t][r];
    }
}

// ---------------------------------------------------------------------------
// OUT[s][e] = (AO @ Wo^T + bo) * layerscale   (fp32 output)
// ---------------------------------------------------------------------------
__global__ __launch_bounds__(256)
void k_gemm_proj(const bf16_t* __restrict__ AO, const bf16_t* __restrict__ W,
                 const float* __restrict__ bias, const float* __restrict__ ls,
                 float* __restrict__ OUT) {
    const int K = DMODEL;
    const int wave = threadIdx.x >> 5;
    const int lane = threadIdx.x & 31;
    const int row0 = blockIdx.x * 16;
    const int n0   = wave * 128;
    const int mb   = (lane & 16) ? 8 : 0;

    bf16_t* panel = stage_a_panel(AO + (size_t)row0 * K, K);

    v8f acc[8];
    zero_acc(acc);
    gemm_strip8(panel, W + (size_t)n0 * K, K, acc);

#pragma unroll
    for (int t = 0; t < 8; ++t) {
        const int n = n0 + t * 16 + (lane & 15);
        const float bn = bias[n], sc = ls[n];
#pragma unroll
        for (int r = 0; r < 8; ++r)
            OUT[(size_t)(row0 + mb + r) * DMODEL + n] = (acc[t][r] + bn) * sc;
    }
}

// ---------------------------------------------------------------------------
// Host-side launcher
// ---------------------------------------------------------------------------
extern "C" void kernel_launch(void* const* d_in, const int* in_sizes, int n_in,
                              void* d_out, int out_size, void* d_ws, size_t ws_size,
                              hipStream_t stream) {
    const float* x  = (const float*)d_in[0];
    const float* Wq = (const float*)d_in[1];
    const float* bq = (const float*)d_in[2];
    const float* Wk = (const float*)d_in[3];
    const float* bk = (const float*)d_in[4];
    const float* Wv = (const float*)d_in[5];
    const float* bv = (const float*)d_in[6];
    const float* Wo = (const float*)d_in[7];
    const float* bo = (const float*)d_in[8];
    const float* ls = (const float*)d_in[9];
    float* out = (float*)d_out;

    // workspace partition (~120 MB total)
    char* ws = (char*)d_ws;
    size_t off = 0;
    auto carve = [&](size_t bytes) {
        void* p = ws + off;
        off += (bytes + 255) & ~(size_t)255;
        return p;
    };
    bf16_t* xbf = (bf16_t*)carve((size_t)BS * DMODEL * 2);
    bf16_t* wqb = (bf16_t*)carve((size_t)DMODEL * DMODEL * 2);
    bf16_t* wkb = (bf16_t*)carve((size_t)DMODEL * DMODEL * 2);
    bf16_t* wvb = (bf16_t*)carve((size_t)DMODEL * DMODEL * 2);
    bf16_t* wob = (bf16_t*)carve((size_t)DMODEL * DMODEL * 2);
    bf16_t* qn  = (bf16_t*)carve((size_t)BS * DMODEL * 2);
    bf16_t* kn  = (bf16_t*)carve((size_t)BS * DMODEL * 2);
    bf16_t* vt  = (bf16_t*)carve((size_t)BATCH * DMODEL * SEQ * 2);
    bf16_t* P   = (bf16_t*)carve((size_t)BS * SEQ * 2);
    bf16_t* ao  = (bf16_t*)carve((size_t)BS * DMODEL * 2);

    const int lds1024 = lds_panel_bytes(DMODEL);  // 33,280 B
    const int lds2048 = lds_panel_bytes(SEQ);     // 66,560 B

    // 1) bf16 conversions
    k_cvt_bf16<<<4096, 256, 0, stream>>>(x,  xbf, BS * DMODEL);
    k_cvt_bf16<<<2048, 256, 0, stream>>>(Wq, wqb, DMODEL * DMODEL);
    k_cvt_bf16<<<2048, 256, 0, stream>>>(Wk, wkb, DMODEL * DMODEL);
    k_cvt_bf16<<<2048, 256, 0, stream>>>(Wv, wvb, DMODEL * DMODEL);
    k_cvt_bf16<<<2048, 256, 0, stream>>>(Wo, wob, DMODEL * DMODEL);

    // 2) projections (+LN for q,k; transposed store for v)
    k_gemm_ln<<<BS / 16, 256, lds1024, stream>>>(xbf, wqb, bq, qn);
    k_gemm_ln<<<BS / 16, 256, lds1024, stream>>>(xbf, wkb, bk, kn);
    k_gemm_vt<<<BS / 16, 256, lds1024, stream>>>(xbf, wvb, bv, vt);

    // 3) sigmoid attention probabilities
    k_gemm_scores<<<dim3(BS / 16, 2), 256, lds1024, stream>>>(qn, kn, P);

    // 4) P @ V
    k_gemm_pv<<<BS / 16, 256, lds2048, stream>>>(P, vt, ao);

    // 5) output projection + layerscale
    k_gemm_proj<<<BS / 16, 256, lds1024, stream>>>(ao, wob, bo, ls, out);
}
